// PlacementGNN_39032662786768
// MI455X (gfx1250) — matrix operand order repaired and checked by reading, not compiled, not run on hardware
//
#include <hip/hip_runtime.h>
#include <hip/hip_bf16.h>

// ---------------------------------------------------------------------------
// PlacementGNN forward for MI455X (gfx1250, wave32, WMMA).
// Dense GEMMs on v_wmma_f32_16x16x32_bf16 (bf16 in, f32 accum).
// Weights are pre-swizzled into per-lane WMMA fragment order so each B
// fragment is two global_load_b128 instead of 16 global_load_u16.
// ---------------------------------------------------------------------------

typedef __attribute__((ext_vector_type(16))) __bf16 v16bf;
typedef __attribute__((ext_vector_type(8)))  float  v8f;

#define WMMA_BF16(a, b, c) \
  __builtin_amdgcn_wmma_f32_16x16x32_bf16(false, (a), false, (b), (short)0, (c), false, false)

__device__ __forceinline__ __bf16 f2bf(float f) {
  unsigned int u = __builtin_bit_cast(unsigned int, f);
  unsigned int r = u + 0x7FFFu + ((u >> 16) & 1u);   // round-to-nearest-even
  unsigned short h = (unsigned short)(r >> 16);
  return __builtin_bit_cast(__bf16, h);
}

// A-matrix (16x32 bf16, MxK) fragment from row-major [16][ldK] LDS buffer.
// Per-lane bytes are two contiguous 16B runs -> 2x ds_load_b128.
__device__ __forceinline__ v16bf load_A(const __bf16* src, int ldK, int k0, int lane) {
  int half = lane >> 4, m = lane & 15;
  v16bf a;
#pragma unroll
  for (int e = 0; e < 16; ++e) {
    int k = k0 + ((e >> 3) << 4) + (half << 3) + (e & 7);
    a[e] = src[m * ldK + k];
  }
  return a;
}

// B fragment from pre-swizzled weights: tile (kc,nt) stored as [32 lanes][16 bf16].
// One aligned 32-byte load -> 2x global_load_b128.
__device__ __forceinline__ v16bf load_Bswz(const __bf16* Wz, int kc, int nt, int NT, int lane) {
  return *reinterpret_cast<const v16bf*>(Wz + ((((kc * NT + nt) << 5) + lane) << 4));
}

__device__ __forceinline__ float sigmoidf_(float x) { return 1.f / (1.f + __expf(-x)); }

// ---------------------------------------------------------------------------
// Helper kernels
// ---------------------------------------------------------------------------
__global__ void fill_kernel(float* p, int n) {
  int i = blockIdx.x * blockDim.x + threadIdx.x;
  if (i < n) p[i] = 0.f;
}

__global__ void count_kernel(const int* __restrict__ dst, float* __restrict__ cnt, int E) {
  int e = blockIdx.x * blockDim.x + threadIdx.x;
  if (e < E) atomicAdd(&cnt[dst[e]], 1.0f);
}

// fp32 [K][N] row-major -> bf16 WMMA-fragment order, zero-padded to Kpad rows.
// Output element o: e=o&15, lane=(o>>4)&31, tile=o>>9 (nt=tile%NT, kc=tile/NT);
// holds W[kc*32 + 16*(e>>3) + 8*(lane>>4) + (e&7)][nt*16 + (lane&15)].
__global__ void conv_swz(const float* __restrict__ src, __bf16* __restrict__ dst,
                         int K, int N, int Kpad) {
  int o = blockIdx.x * blockDim.x + threadIdx.x;
  if (o >= Kpad * N) return;
  int NT = N >> 4;
  int e = o & 15;
  int lane = (o >> 4) & 31;
  int tile = o >> 9;
  int nt = tile % NT, kc = tile / NT;
  int k = kc * 32 + ((e >> 3) << 4) + ((lane >> 4) << 3) + (e & 7);
  int n = nt * 16 + (lane & 15);
  dst[o] = f2bf(k < K ? src[k * N + n] : 0.f);
}

// Feature projection (K=8 or 4, trivially memory-bound -> VALU).
__global__ void proj_kernel(const float* __restrict__ X, const float* __restrict__ W,
                            const float* __restrict__ b, float* __restrict__ H,
                            int rows, int K) {
  int idx = blockIdx.x * blockDim.x + threadIdx.x;
  if (idx >= rows * 64) return;
  int r = idx >> 6, n = idx & 63;
  float acc = b[n];
  for (int k = 0; k < K; ++k) acc += X[r * K + k] * W[k * 64 + n];
  H[idx] = acc;
}

// ---------------------------------------------------------------------------
// Edge message MLP + atomic scatter.  1 wave = 32 edges (two 16-row M-tiles
// sharing each B fragment).  mi = [hA[src]; hB[dst]; attr] (K=131 pad 160)
// h = relu(LN(mi@W1+b1)) @ W2 + b2 ;  agg[dst] += h
// ---------------------------------------------------------------------------
__global__ __launch_bounds__(32)
void msg_kernel(const float* __restrict__ hA, const float* __restrict__ hB,
                const int* __restrict__ src, const int* __restrict__ dst,
                const float* __restrict__ attr,
                const __bf16* __restrict__ W1z, const float* __restrict__ b1,
                const float* __restrict__ lng, const float* __restrict__ lnb,
                const __bf16* __restrict__ W2z, const float* __restrict__ b2,
                float* __restrict__ agg, int E) {
  __shared__ __bf16 lA[32 * 160];
  __shared__ float  lC[32 * 64];
  __shared__ __bf16 lA2[32 * 64];
  __shared__ int    lidx[32];

  int t = threadIdx.x, lane = t, half = lane >> 4;
  int e0 = blockIdx.x * 32;

  { // stage edge indices once
    int e = e0 + t;
    lidx[t] = (e < E) ? ((t < 32) ? 0 : 0) : 0;  // placeholder overwritten below
    int s = 0, d = 0;
    if (e < E) { s = src[e]; d = dst[e]; }
    lidx[t] = s;              // src in lidx
    __syncthreads();
    // gather: lanes 0-15 fetch src row (float4 each), lanes 16-31 dst row
    for (int r = 0; r < 32; ++r) {
      int e2 = e0 + r;
      float4 v = {0.f, 0.f, 0.f, 0.f};
      if (e2 < E) {
        if (t < 16) v = ((const float4*)hA)[lidx[r] * 16 + t];
        else        v = ((const float4*)hB)[dst[e2] * 16 + (t - 16)];
      }
      int k0 = (t < 16) ? (t * 4) : (64 + (t - 16) * 4);
      __bf16* row = lA + r * 160 + k0;
      row[0] = f2bf(v.x); row[1] = f2bf(v.y); row[2] = f2bf(v.z); row[3] = f2bf(v.w);
    }
    // attr + zero pad (k = 128..159)
    for (int i = t; i < 32 * 32; i += 32) {
      int r = i >> 5, k = 128 + (i & 31);
      int e2 = e0 + r;
      float v = (e2 < E && k < 131) ? attr[e2 * 3 + (k - 128)] : 0.f;
      lA[r * 160 + k] = f2bf(v);
    }
    __syncthreads();
    lidx[t] = d;  // now keep dst for the scatter stage (src no longer needed)
  }

  // lin1: [32x160]@[160x64]; B fragment shared by both M-tiles.
#pragma unroll
  for (int nt = 0; nt < 4; ++nt) {
    v8f acc0 = {}, acc1 = {};
#pragma unroll
    for (int kc = 0; kc < 5; ++kc) {
      v16bf b  = load_Bswz(W1z, kc, nt, 4, lane);
      v16bf a0 = load_A(lA,            160, kc * 32, lane);
      v16bf a1 = load_A(lA + 16 * 160, 160, kc * 32, lane);
      acc0 = WMMA_BF16(a0, b, acc0);
      acc1 = WMMA_BF16(a1, b, acc1);
    }
    int n = lane & 15;
    float bias = b1[nt * 16 + n];
#pragma unroll
    for (int i = 0; i < 8; ++i) {
      int m = i + 8 * half;
      lC[m * 64 + nt * 16 + n]        = acc0[i] + bias;
      lC[(16 + m) * 64 + nt * 16 + n] = acc1[i] + bias;
    }
  }
  __syncthreads();

  // Per-row LayerNorm + ReLU (one row per lane), re-quantize to bf16.
  {
    float s = 0.f;
    for (int k = 0; k < 64; ++k) s += lC[t * 64 + k];
    float mean = s * (1.f / 64.f);
    float vs = 0.f;
    for (int k = 0; k < 64; ++k) { float d = lC[t * 64 + k] - mean; vs += d * d; }
    float inv = rsqrtf(vs * (1.f / 64.f) + 1e-5f);
    for (int k = 0; k < 64; ++k) {
      float v = (lC[t * 64 + k] - mean) * inv * lng[k] + lnb[k];
      lA2[t * 64 + k] = f2bf(v > 0.f ? v : 0.f);
    }
  }
  __syncthreads();

  // lin2: [32x64]@[64x64] + atomic scatter (bias folds through segment mean).
#pragma unroll
  for (int nt = 0; nt < 4; ++nt) {
    v8f acc0 = {}, acc1 = {};
#pragma unroll
    for (int kc = 0; kc < 2; ++kc) {
      v16bf b  = load_Bswz(W2z, kc, nt, 4, lane);
      v16bf a0 = load_A(lA2,           64, kc * 32, lane);
      v16bf a1 = load_A(lA2 + 16 * 64, 64, kc * 32, lane);
      acc0 = WMMA_BF16(a0, b, acc0);
      acc1 = WMMA_BF16(a1, b, acc1);
    }
    int n = lane & 15;
    float bias = b2[nt * 16 + n];
#pragma unroll
    for (int i = 0; i < 8; ++i) {
      int m = i + 8 * half;
      if (e0 + m < E)
        atomicAdd(&agg[lidx[m] * 64 + nt * 16 + n], acc0[i] + bias);
      if (e0 + 16 + m < E)
        atomicAdd(&agg[lidx[16 + m] * 64 + nt * 16 + n], acc1[i] + bias);
    }
  }
}

// ---------------------------------------------------------------------------
// GRU update + LayerNorm.  1 wave = 16 rows.  Gate columns k, 64+k, 128+k all
// map to the same lane (same k mod 16), so gates are computed directly on the
// six WMMA accumulators in registers -- no big LDS round-trip.
// ---------------------------------------------------------------------------
__global__ __launch_bounds__(32)
void gru_kernel(float* __restrict__ h, const float* __restrict__ agg,
                const float* __restrict__ cnt,
                const __bf16* __restrict__ Wihz, const __bf16* __restrict__ Whhz,
                const float* __restrict__ bih, const float* __restrict__ bhh,
                const float* __restrict__ lng, const float* __restrict__ lnb, int N) {
  __shared__ __bf16 lX[16 * 64];
  __shared__ __bf16 lH[16 * 64];
  __shared__ float  lHf[16 * 64];
  __shared__ float  lT[16 * 64];

  int t = threadIdx.x, lane = t, half = lane >> 4;
  int r0 = blockIdx.x * 16;

  // Gather: two rows per iteration; 16 lanes x float4 per row.
  for (int rr = 0; rr < 16; rr += 2) {
    int r = rr + half;
    int row = r0 + r;
    int p = lane & 15;
    float4 x = {0.f, 0.f, 0.f, 0.f}, hh = {0.f, 0.f, 0.f, 0.f};
    if (row < N) {
      float c = cnt[row]; c = c > 1.f ? c : 1.f;
      float4 a4 = ((const float4*)agg)[row * 16 + p];
      x.x = a4.x / c; x.y = a4.y / c; x.z = a4.z / c; x.w = a4.w / c;
      hh = ((const float4*)h)[row * 16 + p];
    }
    int o = r * 64 + p * 4;
    lX[o] = f2bf(x.x); lX[o + 1] = f2bf(x.y); lX[o + 2] = f2bf(x.z); lX[o + 3] = f2bf(x.w);
    lH[o] = f2bf(hh.x); lH[o + 1] = f2bf(hh.y); lH[o + 2] = f2bf(hh.z); lH[o + 3] = f2bf(hh.w);
    lHf[o] = hh.x; lHf[o + 1] = hh.y; lHf[o + 2] = hh.z; lHf[o + 3] = hh.w;
  }
  __syncthreads();

  // kt = k-residue group; columns k = kt*16 + (lane&15).
#pragma unroll
  for (int kt = 0; kt < 4; ++kt) {
    v8f xr = {}, xz = {}, xn = {}, hr = {}, hz = {}, hn = {};
#pragma unroll
    for (int kc = 0; kc < 2; ++kc) {
      v16bf ax = load_A(lX, 64, kc * 32, lane);
      xr = WMMA_BF16(ax, load_Bswz(Wihz, kc, kt,     12, lane), xr);
      xz = WMMA_BF16(ax, load_Bswz(Wihz, kc, 4 + kt, 12, lane), xz);
      xn = WMMA_BF16(ax, load_Bswz(Wihz, kc, 8 + kt, 12, lane), xn);
      v16bf ah = load_A(lH, 64, kc * 32, lane);
      hr = WMMA_BF16(ah, load_Bswz(Whhz, kc, kt,     12, lane), hr);
      hz = WMMA_BF16(ah, load_Bswz(Whhz, kc, 4 + kt, 12, lane), hz);
      hn = WMMA_BF16(ah, load_Bswz(Whhz, kc, 8 + kt, 12, lane), hn);
    }
    int n = lane & 15;
    int k = kt * 16 + n;
    float bxr = bih[k], bxz = bih[64 + k], bxn = bih[128 + k];
    float bhr = bhh[k], bhz = bhh[64 + k], bhn = bhh[128 + k];
#pragma unroll
    for (int i = 0; i < 8; ++i) {
      int m = i + 8 * half;
      float rg = sigmoidf_((xr[i] + bxr) + (hr[i] + bhr));
      float zg = sigmoidf_((xz[i] + bxz) + (hz[i] + bhz));
      float ng = tanhf((xn[i] + bxn) + rg * (hn[i] + bhn));
      float ho = lHf[m * 64 + k];
      lT[m * 64 + k] = (1.f - zg) * ng + zg * ho;
    }
  }
  __syncthreads();

  if (t < 16) {
    int row = r0 + t;
    if (row < N) {
      float s = 0.f, ss = 0.f;
      for (int k = 0; k < 64; ++k) { float v = lT[t * 64 + k]; s += v; ss += v * v; }
      float mean = s * (1.f / 64.f);
      float var  = ss * (1.f / 64.f) - mean * mean;
      float inv  = rsqrtf(var + 1e-5f);
      for (int k = 0; k < 64; ++k)
        h[row * 64 + k] = (lT[t * 64 + k] - mean) * inv * lng[k] + lnb[k];
    }
  }
}

// ---------------------------------------------------------------------------
// Head: relu(h@W1+b1) @ W2(64x2) + b2 -> sigmoid placement.  1 wave = 16 rows.
// ---------------------------------------------------------------------------
__global__ __launch_bounds__(32)
void head_kernel(const float* __restrict__ hm, const __bf16* __restrict__ W1z,
                 const float* __restrict__ b1, const float* __restrict__ W2f,
                 const float* __restrict__ b2, const float* __restrict__ msize,
                 const float* __restrict__ cw, const float* __restrict__ ch,
                 float* __restrict__ out, int NMr) {
  __shared__ __bf16 lX[16 * 64];
  __shared__ float  lH[16 * 64];
  int t = threadIdx.x, lane = t, half = lane >> 4;
  int r0 = blockIdx.x * 16;

  for (int rr = 0; rr < 16; rr += 2) {
    int r = rr + half;
    int row = r0 + r;
    int p = lane & 15;
    float4 v = {0.f, 0.f, 0.f, 0.f};
    if (row < NMr) v = ((const float4*)hm)[row * 16 + p];
    int o = r * 64 + p * 4;
    lX[o] = f2bf(v.x); lX[o + 1] = f2bf(v.y); lX[o + 2] = f2bf(v.z); lX[o + 3] = f2bf(v.w);
  }
  __syncthreads();

#pragma unroll
  for (int nt = 0; nt < 4; ++nt) {
    v8f acc = {};
#pragma unroll
    for (int kc = 0; kc < 2; ++kc) {
      v16bf a = load_A(lX, 64, kc * 32, lane);
      v16bf b = load_Bswz(W1z, kc, nt, 4, lane);
      acc = WMMA_BF16(a, b, acc);
    }
    int n = lane & 15;
    float bias = b1[nt * 16 + n];
#pragma unroll
    for (int i = 0; i < 8; ++i) {
      int m = i + 8 * half;
      float v = acc[i] + bias;
      lH[m * 64 + nt * 16 + n] = v > 0.f ? v : 0.f;
    }
  }
  __syncthreads();

  if (t < 16) {
    int row = r0 + t;
    if (row < NMr) {
      float r0v = b2[0], r1v = b2[1];
      for (int k = 0; k < 64; ++k) {
        float hv = lH[t * 64 + k];
        r0v += hv * W2f[k * 2 + 0];
        r1v += hv * W2f[k * 2 + 1];
      }
      float s0 = sigmoidf_(r0v), s1 = sigmoidf_(r1v);
      float m0 = msize[row * 2 + 0], m1 = msize[row * 2 + 1];
      float rx = cw[0] - m0; rx = rx > 1.f ? rx : 1.f;
      float ry = ch[0] - m1; ry = ry > 1.f ? ry : 1.f;
      out[row * 2 + 0] = s0 * rx + 0.5f * m0;
      out[row * 2 + 1] = s1 * ry + 0.5f * m1;
    }
  }
}

// ---------------------------------------------------------------------------
// Host orchestration.  Params (d_in[0..153]) JAX-pytree order:
//   0..3: head.lin1{b,w}, head.lin2{b,w}
//   4+24l: gru_macro{bhh,bih,whh,wih}, gru_net{bhh,bih,whh,wih},
//          ln_macro{b,g}, ln_net{b,g}, m2n{l1b,l1w,l2b,l2w,lnb,lng}, n2m{same}
//   148..153: proj_macro{b,w}, proj_net{b,w}, proj_port{b,w}
// Then macro_f, net_f, port_f, ei_m2n, ea_m2n, ei_n2m, ea_n2m, msize, cw, ch.
// ---------------------------------------------------------------------------
extern "C" void kernel_launch(void* const* d_in, const int* in_sizes, int n_in,
                              void* d_out, int out_size, void* d_ws, size_t ws_size,
                              hipStream_t stream) {
  (void)in_sizes; (void)n_in; (void)out_size; (void)ws_size;
  const int NM = 50000, NP = 10000, NN_ = 60000, NNODES = 60000, E = 150000, NL = 6;

  auto pf = [&](int i) { return (const float*)d_in[i]; };
  auto pi = [&](int i) { return (const int*)d_in[i]; };

  const int IB = 4 + 24 * NL + 6;  // 154
  const float* macro_f = pf(IB + 0);
  const float* net_f   = pf(IB + 1);
  const float* port_f  = pf(IB + 2);
  const int*   ei_m2n  = pi(IB + 3);
  const float* ea_m2n  = pf(IB + 4);
  const int*   ei_n2m  = pi(IB + 5);
  const float* ea_n2m  = pf(IB + 6);
  const float* msize   = pf(IB + 7);
  const float* cw      = pf(IB + 8);
  const float* ch      = pf(IB + 9);

  float* h_nodes = (float*)d_ws;                 // NNODES*64
  float* h_net   = h_nodes + NNODES * 64;
  float* agg     = h_net + NN_ * 64;
  float* cnt_net = agg + NNODES * 64;
  float* cnt_nod = cnt_net + NN_;
  __bf16* w1a  = (__bf16*)(cnt_nod + NNODES);    // 160*64 (swizzled)
  __bf16* w2a  = w1a + 160 * 64;                 // 64*64
  __bf16* w1b  = w2a + 64 * 64;
  __bf16* w2b  = w1b + 160 * 64;
  __bf16* wihn = w2b + 64 * 64;                  // 64*192 each
  __bf16* whhn = wihn + 64 * 192;
  __bf16* wihm = whhn + 64 * 192;
  __bf16* whhm = wihm + 64 * 192;
  __bf16* w1h  = whhm + 64 * 192;                // 64*64

  const int TB = 256;
  auto cdiv = [](int a, int b) { return (a + b - 1) / b; };

  // Segment counts are layer-invariant: compute once.
  fill_kernel<<<cdiv(NN_, TB), TB, 0, stream>>>(cnt_net, NN_);
  fill_kernel<<<cdiv(NNODES, TB), TB, 0, stream>>>(cnt_nod, NNODES);
  count_kernel<<<cdiv(E, TB), TB, 0, stream>>>(ei_m2n + E, cnt_net, E);
  count_kernel<<<cdiv(E, TB), TB, 0, stream>>>(ei_n2m + E, cnt_nod, E);

  // Feature projections (h_nodes = [macros ; ports]).
  proj_kernel<<<cdiv(NM * 64, TB), TB, 0, stream>>>(macro_f, pf(149), pf(148), h_nodes, NM, 8);
  proj_kernel<<<cdiv(NP * 64, TB), TB, 0, stream>>>(port_f, pf(153), pf(152),
                                                    h_nodes + NM * 64, NP, 4);
  proj_kernel<<<cdiv(NN_ * 64, TB), TB, 0, stream>>>(net_f, pf(151), pf(150), h_net, NN_, 4);

  for (int l = 0; l < NL; ++l) {
    int B0 = 4 + 24 * l;
    conv_swz<<<cdiv(160 * 64, TB), TB, 0, stream>>>(pf(B0 + 13), w1a, 131, 64, 160);
    conv_swz<<<cdiv(64 * 64, TB), TB, 0, stream>>>(pf(B0 + 15), w2a, 64, 64, 64);
    conv_swz<<<cdiv(160 * 64, TB), TB, 0, stream>>>(pf(B0 + 19), w1b, 131, 64, 160);
    conv_swz<<<cdiv(64 * 64, TB), TB, 0, stream>>>(pf(B0 + 21), w2b, 64, 64, 64);
    conv_swz<<<cdiv(64 * 192, TB), TB, 0, stream>>>(pf(B0 + 7), wihn, 64, 192, 64);
    conv_swz<<<cdiv(64 * 192, TB), TB, 0, stream>>>(pf(B0 + 6), whhn, 64, 192, 64);
    conv_swz<<<cdiv(64 * 192, TB), TB, 0, stream>>>(pf(B0 + 3), wihm, 64, 192, 64);
    conv_swz<<<cdiv(64 * 192, TB), TB, 0, stream>>>(pf(B0 + 2), whhm, 64, 192, 64);

    // macro/port -> net messages, then net GRU.
    fill_kernel<<<cdiv(NN_ * 64, TB), TB, 0, stream>>>(agg, NN_ * 64);
    msg_kernel<<<cdiv(E, 32), 32, 0, stream>>>(
        h_nodes, h_net, ei_m2n, ei_m2n + E, ea_m2n,
        w1a, pf(B0 + 12), pf(B0 + 17), pf(B0 + 16), w2a, pf(B0 + 14), agg, E);
    gru_kernel<<<cdiv(NN_, 16), 32, 0, stream>>>(
        h_net, agg, cnt_net, wihn, whhn, pf(B0 + 5), pf(B0 + 4),
        pf(B0 + 11), pf(B0 + 10), NN_);

    // net -> macro/port messages, then node GRU.
    fill_kernel<<<cdiv(NNODES * 64, TB), TB, 0, stream>>>(agg, NNODES * 64);
    msg_kernel<<<cdiv(E, 32), 32, 0, stream>>>(
        h_net, h_nodes, ei_n2m, ei_n2m + E, ea_n2m,
        w1b, pf(B0 + 18), pf(B0 + 23), pf(B0 + 22), w2b, pf(B0 + 20), agg, E);
    gru_kernel<<<cdiv(NNODES, 16), 32, 0, stream>>>(
        h_nodes, agg, cnt_nod, wihm, whhm, pf(B0 + 1), pf(B0 + 0),
        pf(B0 + 9), pf(B0 + 8), NNODES);
  }

  conv_swz<<<cdiv(64 * 64, TB), TB, 0, stream>>>(pf(1), w1h, 64, 64, 64);
  head_kernel<<<cdiv(NM, 16), 32, 0, stream>>>(
      h_nodes, w1h, pf(0), pf(3), pf(2), msize, cw, ch, (float*)d_out, NM);
}
